// VGG16RoIHead_14448269983854
// MI455X (gfx1250) — matrix operand order repaired and checked by reading, not compiled, not run on hardware
//
#include <hip/hip_runtime.h>

// ---------------------------------------------------------------------------
// VGG16 RoI head for MI455X (gfx1250, wave32, WMMA + Tensor Data Mover).
//
// Roofline: fc6/fc7 are weight-streaming GEMMs (478 MB fp32 weights @23.3TB/s
// ~= 20us floor). AI = 64 FLOP/byte -> fp32 WMMA (16x16x4) is ~8x too weak,
// so weights are converted fp32->bf16 in registers and fed to
// v_wmma_f32_16x16x32_bf16. To make the weight stream cross HBM once, each
// workgroup (4 waves, 64 output rows) stages its 32x32 fp32 weight tile in
// LDS via the TDM (TENSOR_LOAD_TO_LDS, double-buffered, s_wait_tensorcnt),
// with D# padding (+1 DWORD per 32-DWORD row -> LDS stride 33) so the
// column-walk reads are bank-conflict free between half-waves.
// ---------------------------------------------------------------------------

typedef __attribute__((ext_vector_type(16))) __bf16 bf16x16;
typedef __attribute__((ext_vector_type(8)))  __bf16 bf16x8;
typedef __attribute__((ext_vector_type(8)))  float  f32x8;
typedef __attribute__((ext_vector_type(4)))  unsigned int u32x4;
typedef __attribute__((ext_vector_type(8)))  int    i32x8;
typedef __attribute__((ext_vector_type(4)))  int    i32x4;

static constexpr int R_   = 128;
static constexpr int C_   = 512;
static constexpr int H_   = 37;
static constexpr int W_   = 37;
static constexpr int PP   = 7;              // pooled size
static constexpr int D_IN = C_ * PP * PP;   // 25088
static constexpr int D_H  = 4096;
static constexpr int NCLS = 21;
static constexpr float SCALE_ = 1.0f / 16.0f;

static constexpr int TILE_K   = 32;
static constexpr int TILE_N   = 32;         // NT=2 x 16
static constexpr int LDS_PITCH = 33;        // 32 + 1 DWORD TDM pad
static constexpr int WAVES    = 4;          // waves per block (4 M-tiles)

// ---------------------------------------------------------------------------
// Kernel 1: RoI max pool -> bf16 activation matrix A [R, D_IN] row-major.
// jnp.round == rintf (round half to even); float clip then int cast.
// ---------------------------------------------------------------------------
__global__ void roi_pool_bf16(const float* __restrict__ x,
                              const float* __restrict__ rois,   // [R,4] = (y1,x1,y2,x2)
                              const int*   __restrict__ ridx,
                              __bf16* __restrict__ poolA) {
  int idx = blockIdx.x * blockDim.x + threadIdx.x;
  if (idx >= R_ * C_ * PP * PP) return;
  int pw = idx % PP;
  int ph = (idx / PP) % PP;
  int c  = (idx / (PP * PP)) % C_;
  int r  = idx / (PP * PP * C_);

  float y1 = rois[r * 4 + 0], x1 = rois[r * 4 + 1];
  float y2 = rois[r * 4 + 2], x2 = rois[r * 4 + 3];
  float swf = rintf(x1 * SCALE_), ewf = rintf(x2 * SCALE_);
  float shf = rintf(y1 * SCALE_), ehf = rintf(y2 * SCALE_);
  float bw = fmaxf(ewf - swf + 1.0f, 1.0f) / (float)PP;
  float bh = fmaxf(ehf - shf + 1.0f, 1.0f) / (float)PP;

  int hs = (int)fminf(fmaxf(floorf((float)ph * bh) + shf, 0.0f), (float)H_);
  int he = (int)fminf(fmaxf(ceilf((float)(ph + 1) * bh) + shf, 0.0f), (float)H_);
  int ws = (int)fminf(fmaxf(floorf((float)pw * bw) + swf, 0.0f), (float)W_);
  int we = (int)fminf(fmaxf(ceilf((float)(pw + 1) * bw) + swf, 0.0f), (float)W_);

  const float* feat = x + (((size_t)ridx[r] * C_ + c) * H_) * W_;
  float m = -INFINITY;
  for (int h = hs; h < he; ++h)
    for (int w = ws; w < we; ++w)
      m = fmaxf(m, feat[h * W_ + w]);
  if (he <= hs || we <= ws) m = 0.0f;

  poolA[(size_t)r * D_IN + c * (PP * PP) + ph * PP + pw] = (__bf16)m;
}

// ---------------------------------------------------------------------------
// TDM: DMA one 32x32 fp32 tile (row stride N floats) from global -> LDS,
// inserting 1 pad DWORD after every 32 DWORDs (LDS row pitch = 33 floats).
// D# layout per CDNA5 ISA.
// ---------------------------------------------------------------------------
__device__ __forceinline__ void tdm_load_tile(const float* gsrc, unsigned lds_off,
                                              int N, int K) {
  u32x4 g0;
  unsigned long long ga = (unsigned long long)(uintptr_t)gsrc;
  g0.x = 1u;                                   // count=1, is_restore=0, gather off
  g0.y = lds_off;                              // lds_addr (bytes)
  g0.z = (unsigned)ga;                         // global_addr[31:0]
  g0.w = (unsigned)(ga >> 32) | (2u << 30);    // global_addr[56:32] | type=2

  i32x8 g1;
  g1[0] = (2 << 16)        // data_size = 4B
        | (1 << 20)        // pad_enable
        | (4 << 22)        // pad_interval: 2<<4 = 32 DWORDs
        | (0 << 25);       // pad_amount: 1 DWORD
  g1[1] = (N & 0xFFFF) << 16;                          // tensor_dim0[15:0]
  g1[2] = ((N >> 16) & 0xFFFF) | ((K & 0xFFFF) << 16); // dim0 hi | tensor_dim1 lo
  g1[3] = ((K >> 16) & 0xFFFF) | (TILE_N << 16);       // dim1 hi | tile_dim0
  g1[4] = TILE_K;                                      // tile_dim1 (tile_dim2=0)
  g1[5] = N;                                           // tensor_dim0_stride[31:0]
  g1[6] = 0;
  g1[7] = 0;

  i32x4 z4 = {0, 0, 0, 0};
#if __has_include(<hip/amd_detail/amd_gfx1250_TDM.h>)
  i32x8 z8 = {0, 0, 0, 0, 0, 0, 0, 0};
  __builtin_amdgcn_tensor_load_to_lds(g0, g1, z4, z4, z8, 0);
#else
  __builtin_amdgcn_tensor_load_to_lds(g0, g1, z4, z4, 0);
#endif
}

// ---------------------------------------------------------------------------
// Kernels 2/3: C = act(A[M,K](bf16) x W[K,N](fp32) + bias), W staged in LDS
// by the TDM, converted to bf16 in-regs, v_wmma_f32_16x16x32_bf16.
//
// Block: 128 threads = 4 waves; each wave owns a 16x32 output tile (all waves
// share the same 32 columns -> each weight tile is DMAed once per block).
//
// ISA VGPR layouts (CDNA5 ISA, wave32):
//  A (16x32 bf16):  lane m = lane&15, half = lane>>4;
//                   elements 0..7  -> K = kb + half*8 + e        (contig 16B)
//                   elements 8..15 -> K = kb + 16 + half*8 + e-8 (contig 16B)
//  B (32x16 bf16):  N = lane&15; as A: e<8 -> K = kb + half*8 + e, e>=8 -> K = kb + 16 + half*8 + e-8
//  C/D (16x16 f32): VGPR j -> M = j + 8*half, N = lane&15
// ---------------------------------------------------------------------------
__global__ void gemm_wmma_tdm(const __bf16* __restrict__ A,   // [M,K] bf16
                              const float*  __restrict__ Wt,  // [K,N] fp32
                              const float*  __restrict__ bias,// [N]
                              __bf16* __restrict__ Cout,      // [M,N] bf16
                              int N, int K, int relu) {
  __shared__ float tile[2][TILE_K * LDS_PITCH];

  const int lane   = threadIdx.x & 31;
  const int waveId = threadIdx.x >> 5;
  const int half   = lane >> 4;
  const int l16    = lane & 15;
  const int mTile  = (blockIdx.y * WAVES + waveId) * 16;
  const int nBase  = blockIdx.x * TILE_N;

  const unsigned lds0 = (unsigned)(uintptr_t)&tile[0][0];
  const unsigned lds1 = (unsigned)(uintptr_t)&tile[1][0];

  f32x8 acc[2];
#pragma unroll
  for (int t = 0; t < 2; ++t)
    acc[t] = f32x8{0.f, 0.f, 0.f, 0.f, 0.f, 0.f, 0.f, 0.f};

  const __bf16* Arow = A + (size_t)(mTile + l16) * K + half * 8;

  // prologue: DMA first weight tile
  if (waveId == 0) tdm_load_tile(Wt + nBase, lds0, N, K);

  int cur = 0;
  for (int kb = 0; kb < K; kb += TILE_K) {
    const bool more = (kb + TILE_K) < K;
    if (waveId == 0) {
      if (more) {
        tdm_load_tile(Wt + (size_t)(kb + TILE_K) * N + nBase,
                      cur ? lds0 : lds1, N, K);
        __builtin_amdgcn_s_wait_tensorcnt((short)1);  // oldest (cur) landed
      } else {
        __builtin_amdgcn_s_wait_tensorcnt((short)0);
      }
    }
    __syncthreads();

    // ---- A fragment: two contiguous 16B bf16 loads per lane ----
    bf16x8 a0 = *(const bf16x8*)(Arow + kb);
    bf16x8 a1 = *(const bf16x8*)(Arow + kb + 16);
    bf16x16 aF;
#pragma unroll
    for (int e = 0; e < 8; ++e) { aF[e] = a0[e]; aF[e + 8] = a1[e]; }

    // ---- B fragments from LDS (pitch 33 -> conflict-free halves) ----
    const float* buf = &tile[cur][(size_t)(8 * half) * LDS_PITCH + l16];
#pragma unroll
    for (int t = 0; t < 2; ++t) {
      bf16x16 bF;
#pragma unroll
      for (int e = 0; e < 16; ++e)
        bF[e] = (__bf16)buf[(e + (e >= 8 ? 8 : 0)) * LDS_PITCH + t * 16];

      acc[t] = __builtin_amdgcn_wmma_f32_16x16x32_bf16(
          /*neg_a=*/false, aF, /*neg_b=*/false, bF,
          /*c_mod=*/(short)0, acc[t], /*reuse_a=*/false, /*reuse_b=*/false);
    }

    __syncthreads();   // done reading tile[cur]; TDM may overwrite it next iter
    cur ^= 1;
  }

  // ---- epilogue: bias + ReLU, store bf16 per C/D layout ----
#pragma unroll
  for (int t = 0; t < 2; ++t) {
    const int ncol = nBase + t * 16 + l16;
    const float bv = bias[ncol];
#pragma unroll
    for (int j = 0; j < 8; ++j) {
      float v = acc[t][j] + bv;
      if (relu) v = fmaxf(v, 0.0f);
      Cout[(size_t)(mTile + j + 8 * half) * N + ncol] = (__bf16)v;
    }
  }
}

// ---------------------------------------------------------------------------
// Kernel 4: tiny heads. out = [fc7 @ w_loc + b_loc (128x84) | fc7 @ w_score +
// b_score (128x21)] concatenated flat. fc7 (1MB bf16) is L2-resident.
// ---------------------------------------------------------------------------
__global__ void heads_kernel(const __bf16* __restrict__ fc7,   // [R, D_H] bf16
                             const float* __restrict__ w_loc,  // [D_H, 84]
                             const float* __restrict__ b_loc,
                             const float* __restrict__ w_sc,   // [D_H, 21]
                             const float* __restrict__ b_sc,
                             float* __restrict__ out) {
  const int NJ = NCLS * 4 + NCLS;   // 105
  int tid = blockIdx.x * blockDim.x + threadIdx.x;
  if (tid >= R_ * NJ) return;
  int r = tid / NJ, j = tid % NJ;

  const float* wcol;
  float bv;
  float* dst;
  int ldw;
  if (j < NCLS * 4) {
    wcol = w_loc + j; ldw = NCLS * 4; bv = b_loc[j];
    dst = out + (size_t)r * (NCLS * 4) + j;
  } else {
    int jj = j - NCLS * 4;
    wcol = w_sc + jj; ldw = NCLS; bv = b_sc[jj];
    dst = out + (size_t)R_ * (NCLS * 4) + (size_t)r * NCLS + jj;
  }

  const __bf16* arow = fc7 + (size_t)r * D_H;
  float s = 0.0f;
#pragma unroll 8
  for (int k = 0; k < D_H; ++k)
    s += (float)arow[k] * wcol[(size_t)k * ldw];
  *dst = s + bv;
}

// ---------------------------------------------------------------------------
extern "C" void kernel_launch(void* const* d_in, const int* in_sizes, int n_in,
                              void* d_out, int out_size, void* d_ws, size_t ws_size,
                              hipStream_t stream) {
  const float* x     = (const float*)d_in[0];
  const float* rois  = (const float*)d_in[1];
  const int*   ridx  = (const int*)  d_in[2];
  const float* w1    = (const float*)d_in[3];
  const float* b1    = (const float*)d_in[4];
  const float* w2    = (const float*)d_in[5];
  const float* b2    = (const float*)d_in[6];
  const float* w_loc = (const float*)d_in[7];
  const float* b_loc = (const float*)d_in[8];
  const float* w_sc  = (const float*)d_in[9];
  const float* b_sc  = (const float*)d_in[10];
  float* out = (float*)d_out;

  // scratch carve (16B aligned): poolA | fc6 | fc7 (~8.5 MB total)
  char* ws = (char*)d_ws;
  __bf16* poolA = (__bf16*)ws;                                        // R * D_IN
  __bf16* fc6   = (__bf16*)(ws + (size_t)R_ * D_IN * 2);              // R * D_H
  __bf16* fc7   = (__bf16*)(ws + (size_t)R_ * D_IN * 2 + (size_t)R_ * D_H * 2);

  {
    int total = R_ * C_ * PP * PP;
    roi_pool_bf16<<<(total + 255) / 256, 256, 0, stream>>>(x, rois, ridx, poolA);
  }
  {
    dim3 g(D_H / TILE_N, R_ / (16 * WAVES));   // (128, 2), 4 waves/block
    gemm_wmma_tdm<<<g, 32 * WAVES, 0, stream>>>(poolA, w1, b1, fc6, D_H, D_IN, 1);
  }
  {
    dim3 g(D_H / TILE_N, R_ / (16 * WAVES));
    gemm_wmma_tdm<<<g, 32 * WAVES, 0, stream>>>(fc6, w2, b2, fc7, D_H, D_H, 1);
  }
  {
    int total = R_ * (NCLS * 4 + NCLS);
    heads_kernel<<<(total + 255) / 256, 256, 0, stream>>>(fc7, w_loc, b_loc, w_sc, b_sc, out);
  }
}
